// GPT2Attention_26371099197458
// MI455X (gfx1250) — matrix operand order repaired and checked by reading, not compiled, not run on hardware
//
#include <hip/hip_runtime.h>

// ---------------------------------------------------------------------------
// GPT-2 attention block for MI455X (gfx1250), bf16 WMMA pipeline.
// B=4, T=2048, C=1024, H=16, D=64. ~140 GFLOP vs ~90MB unique traffic ->
// compute bound -> all three GEMM stages + both attention matmuls run on
// v_wmma_f32_16x16x32_bf16. Global->LDS staging is software-pipelined
// (register prefetch + ping-pong LDS) to hide HBM/L2 latency behind WMMAs.
// ---------------------------------------------------------------------------

typedef __attribute__((ext_vector_type(16))) __bf16 v16bf;
typedef __attribute__((ext_vector_type(8)))  float  v8f;

struct __align__(16) U4 { unsigned int x, y, z, w; };

union Frag {            // 8 VGPRs = 16 bf16 = one WMMA A/B operand
  U4    u[2];
  v16bf v;
};

__device__ __forceinline__ unsigned short f2bf(float f) {
  unsigned int u = __float_as_uint(f);
  u += 0x7FFFu + ((u >> 16) & 1u);   // round-to-nearest-even
  return (unsigned short)(u >> 16);
}

__device__ __forceinline__ v8f wmma_bf16(const Frag& a, const Frag& b, v8f c) {
  return __builtin_amdgcn_wmma_f32_16x16x32_bf16(
      false, a.v, false, b.v, (short)0, c, false, false);
}

// ---------------------------------------------------------------------------
// fp32 -> bf16 elementwise convert (all sizes are multiples of 4)
// ---------------------------------------------------------------------------
__global__ __launch_bounds__(256) void cvt_kernel(const float* __restrict__ src,
                                                  unsigned short* __restrict__ dst,
                                                  int n4) {
  int i = blockIdx.x * 256 + threadIdx.x;
  if (i < n4) {
    float4 v = ((const float4*)src)[i];
    uint2 pk;
    pk.x = (unsigned)f2bf(v.x) | ((unsigned)f2bf(v.y) << 16);
    pk.y = (unsigned)f2bf(v.z) | ((unsigned)f2bf(v.w) << 16);
    ((uint2*)dst)[i] = pk;
  }
}

// ---------------------------------------------------------------------------
// GEMM: out[M,N] = Abf16[M,K] @ Wbf16[K,N] + bias.
// Block tile 128x128, BK=32, 256 threads = 8 waves in a 2(M)x4(N) grid,
// each wave computes 4x2 WMMA 16x16 tiles. Ping-pong LDS + register
// prefetch: next K-tile's global loads are in flight during current WMMAs.
// MODE 0: f32 output (projection). MODE 1: QKV epilogue -> Q(scaled)/K bf16
// [B,H,T,D] and V transposed bf16 [B,H,D,T].
// ---------------------------------------------------------------------------
#define APADK 40
#define BPADK 40

template <int MODE>
__global__ __launch_bounds__(256) void gemm_kernel(
    const unsigned short* __restrict__ A, const unsigned short* __restrict__ W,
    const float* __restrict__ bias, int M, int N, int K,
    unsigned short* __restrict__ qb, unsigned short* __restrict__ kbf,
    unsigned short* __restrict__ vtb, float* __restrict__ outf) {
  __shared__ unsigned short Alds[2][128 * APADK];  // [m][k]
  __shared__ unsigned short Blds[2][128 * BPADK];  // [n][k] (transposed on fill)

  const int tid   = threadIdx.x;
  const int lane  = tid & 31;
  const int wave  = tid >> 5;
  const int waveM = wave >> 2;      // 0..1
  const int waveN = wave & 3;       // 0..3
  const int lrow  = lane & 15;
  const int hi    = lane >> 4;      // 0 or 1 (lane half)

  const int bm = blockIdx.y * 128;
  const int bn = blockIdx.x * 128;

  const v8f vzero = {0.f,0.f,0.f,0.f,0.f,0.f,0.f,0.f};
  v8f acc[4][2];
  for (int i = 0; i < 4; ++i)
    for (int j = 0; j < 2; ++j) acc[i][j] = vzero;

  const int arow = tid >> 1;           // 0..127
  const int acol = (tid & 1) * 16;     // 0 or 16
  const int brow = tid >> 3;           // 0..31 (k)
  const int bcol = (tid & 7) * 16;     // 0..112 (n)

  const unsigned short* Abase = A + (size_t)(bm + arow) * K + acol;
  const unsigned short* Wbase = W + (size_t)brow * N + bn + bcol;

  U4 pa0, pa1, pb0, pb1;               // prefetch registers
  auto load_regs = [&](int kk) {
    const U4* s = (const U4*)(Abase + kk);
    pa0 = s[0];
    pa1 = s[1];
    const U4* t = (const U4*)(Wbase + (size_t)kk * N);
    pb0 = t[0];
    pb1 = t[1];
  };
  auto store_lds = [&](int buf) {
    U4* ad = (U4*)&Alds[buf][arow * APADK + acol];
    ad[0] = pa0;
    ad[1] = pa1;
    const unsigned short* s0 = (const unsigned short*)&pb0;
    const unsigned short* s1 = (const unsigned short*)&pb1;
#pragma unroll
    for (int i = 0; i < 8; ++i) {
      Blds[buf][(bcol + i)     * BPADK + brow] = s0[i];
      Blds[buf][(bcol + 8 + i) * BPADK + brow] = s1[i];
    }
  };

  load_regs(0);
  store_lds(0);
  __syncthreads();

  const int nk = K >> 5;
  for (int ki = 0; ki < nk; ++ki) {
    const int cur = ki & 1;
    const bool has_next = (ki + 1) < nk;
    if (has_next) load_regs((ki + 1) << 5);   // global loads in flight...

    Frag af[4], bf[2];
    const int akb = hi ? 8 : 0;     // A: lanes>=16 start at K=8 (and +16)
#pragma unroll
    for (int mt = 0; mt < 4; ++mt) {
      const U4* p = (const U4*)&Alds[cur][(waveM * 64 + mt * 16 + lrow) * APADK + akb];
      af[mt].u[0] = p[0];           // K = akb .. akb+7
      af[mt].u[1] = p[2];           // K = akb+16 .. akb+23
    }
    const int bkb = hi ? 8 : 0;     // B: lanes>=16 start at K=8 (and +16)
#pragma unroll
    for (int nt = 0; nt < 2; ++nt) {
      const U4* p = (const U4*)&Blds[cur][(waveN * 32 + nt * 16 + lrow) * BPADK + bkb];
      bf[nt].u[0] = p[0];
      bf[nt].u[1] = p[2];
    }
#pragma unroll
    for (int mt = 0; mt < 4; ++mt)
#pragma unroll
      for (int nt = 0; nt < 2; ++nt)
        acc[mt][nt] = wmma_bf16(af[mt], bf[nt], acc[mt][nt]);  // ...during these

    if (has_next) {
      store_lds(1 - cur);
      __syncthreads();
    }
  }

  // epilogue. C layout: col = lane%16, row_in_tile = r + 8*hi.
#pragma unroll
  for (int mt = 0; mt < 4; ++mt) {
#pragma unroll
    for (int nt = 0; nt < 2; ++nt) {
      const int col = bn + waveN * 32 + nt * 16 + lrow;
      const float bv = bias[col];
#pragma unroll
      for (int r = 0; r < 8; ++r) {
        const int row = bm + waveM * 64 + mt * 16 + r + hi * 8;
        const float val = acc[mt][nt][r] + bv;
        if constexpr (MODE == 0) {
          outf[(size_t)row * N + col] = val;
        } else {
          const int sec = col >> 10;         // 0=Q 1=K 2=V
          const int cc  = col & 1023;
          const int h   = cc >> 6, d = cc & 63;
          const int b   = row >> 11, t = row & 2047;
          const size_t bh = (size_t)(b * 16 + h);
          if (sec == 0)        // Q scaled by (1/sqrt(64)) * log2(e) for exp2 softmax
            qb[(bh * 2048 + t) * 64 + d] = f2bf(val * 0.18033688011112042f);
          else if (sec == 1)
            kbf[(bh * 2048 + t) * 64 + d] = f2bf(val);
          else                 // V stored transposed [bh][d][t]
            vtb[(bh * 64 + d) * 2048 + t] = f2bf(val);
        }
      }
    }
  }
}

// ---------------------------------------------------------------------------
// Causal flash attention. Block = 4 waves = 64 queries; loop over 64-key
// blocks <= query block. S and O via WMMA; online softmax in exp2 domain.
// K/V tiles ping-pong through LDS with register prefetch; the per-wave
// P transpose uses a wave-local s_wait_dscnt instead of a block barrier
// (LDS ops from one wave complete in order).
// Output y -> bf16 [B,T,C] for the projection GEMM.
// ---------------------------------------------------------------------------
#define KPAD 72

__global__ __launch_bounds__(128) void fa_kernel(
    const unsigned short* __restrict__ Qb, const unsigned short* __restrict__ Kb,
    const unsigned short* __restrict__ Vtb, unsigned short* __restrict__ Yb) {
  __shared__ unsigned short Klds[2][64 * KPAD];    // [key][d]
  __shared__ unsigned short Vlds[2][64 * KPAD];    // [d][key]
  __shared__ unsigned short Plds[4][16 * KPAD];    // per-wave [qrow][key]

  const int tid  = threadIdx.x;
  const int lane = tid & 31;
  const int wave = tid >> 5;
  const int lrow = lane & 15;
  const int hi   = lane >> 4;

  const int qblk  = blockIdx.x;
  const int bh    = blockIdx.y;
  const int qbase = qblk * 64;

  const unsigned short* Qh = Qb  + (size_t)bh * 2048 * 64;
  const unsigned short* Kh = Kb  + (size_t)bh * 2048 * 64;
  const unsigned short* Vh = Vtb + (size_t)bh * 64 * 2048;

  // Q A-fragments, 2 d-chunks of 32, loaded straight from global
  Frag qf[2];
  {
    const int qrow = qbase + wave * 16 + lrow;
#pragma unroll
    for (int c = 0; c < 2; ++c) {
      const U4* p = (const U4*)(Qh + (size_t)qrow * 64 + c * 32 + hi * 8);
      qf[c].u[0] = p[0];
      qf[c].u[1] = p[2];
    }
  }

  const v8f vzero = {0.f,0.f,0.f,0.f,0.f,0.f,0.f,0.f};
  v8f oacc[4];
  for (int i = 0; i < 4; ++i) oacc[i] = vzero;
  float mstate[8], lstate[8];
#pragma unroll
  for (int r = 0; r < 8; ++r) { mstate[r] = -__builtin_inff(); lstate[r] = 0.f; }

  const int qrow_base = qbase + wave * 16 + hi * 8;

  // K/V staging: 64B per thread each, prefetched one key-block ahead
  const int r_st  = tid >> 1;
  const int half  = (tid & 1) * 32;
  const unsigned short* Ksrc = Kh + (size_t)r_st * 64 + half;
  const unsigned short* Vsrc = Vh + (size_t)r_st * 2048 + half;
  U4 kr[4], vr[4];
  auto load_kv = [&](int kbase) {
    const U4* ks = (const U4*)(Ksrc + (size_t)kbase * 64);
    kr[0] = ks[0]; kr[1] = ks[1]; kr[2] = ks[2]; kr[3] = ks[3];
    const U4* vs = (const U4*)(Vsrc + kbase);
    vr[0] = vs[0]; vr[1] = vs[1]; vr[2] = vs[2]; vr[3] = vs[3];
  };
  auto store_kv = [&](int buf) {
    U4* kd = (U4*)&Klds[buf][r_st * KPAD + half];
    kd[0] = kr[0]; kd[1] = kr[1]; kd[2] = kr[2]; kd[3] = kr[3];
    U4* vd = (U4*)&Vlds[buf][r_st * KPAD + half];
    vd[0] = vr[0]; vd[1] = vr[1]; vd[2] = vr[2]; vd[3] = vr[3];
  };

  load_kv(0);
  store_kv(0);
  __syncthreads();

  for (int kb2 = 0; kb2 <= qblk; ++kb2) {
    const int kbase = kb2 * 64;
    const int cur   = kb2 & 1;
    const bool has_next = kb2 < qblk;
    if (has_next) load_kv(kbase + 64);   // next block's global loads in flight

    // S = Q @ K^T : 4 key tiles x 2 d-chunks
    v8f s[4];
#pragma unroll
    for (int kt = 0; kt < 4; ++kt) {
      Frag kf0, kf1;
      const int krow = (kt * 16 + lrow) * KPAD;
      const int dkb  = hi * 8;
      const U4* p0 = (const U4*)&Klds[cur][krow + dkb];
      kf0.u[0] = p0[0]; kf0.u[1] = p0[2];
      const U4* p1 = (const U4*)&Klds[cur][krow + 32 + dkb];
      kf1.u[0] = p1[0]; kf1.u[1] = p1[2];
      v8f z = wmma_bf16(qf[0], kf0, vzero);
      s[kt]  = wmma_bf16(qf[1], kf1, z);
    }

    // causal mask
#pragma unroll
    for (int kt = 0; kt < 4; ++kt) {
      const int key = kbase + kt * 16 + lrow;
#pragma unroll
      for (int r = 0; r < 8; ++r)
        if (key > qrow_base + r) s[kt][r] = -__builtin_inff();
    }

    // online softmax (exp2 domain; scale folded into Q)
    float mnew[8], alpha[8], lad[8];
#pragma unroll
    for (int r = 0; r < 8; ++r) {
      float mb = fmaxf(fmaxf(s[0][r], s[1][r]), fmaxf(s[2][r], s[3][r]));
#pragma unroll
      for (int d = 1; d < 16; d <<= 1) mb = fmaxf(mb, __shfl_xor(mb, d, 32));
      mnew[r] = fmaxf(mstate[r], mb);
      lad[r] = 0.f;
    }
#pragma unroll
    for (int kt = 0; kt < 4; ++kt)
#pragma unroll
      for (int r = 0; r < 8; ++r) {
        const float p = exp2f(s[kt][r] - mnew[r]);
        s[kt][r] = p;
        lad[r] += p;
      }
#pragma unroll
    for (int r = 0; r < 8; ++r) {
      float v = lad[r];
#pragma unroll
      for (int d = 1; d < 16; d <<= 1) v += __shfl_xor(v, d, 32);
      alpha[r]  = exp2f(mstate[r] - mnew[r]);
      lstate[r] = lstate[r] * alpha[r] + v;
      mstate[r] = mnew[r];
    }
#pragma unroll
    for (int dt = 0; dt < 4; ++dt)
#pragma unroll
      for (int r = 0; r < 8; ++r) oacc[dt][r] *= alpha[r];

    // C-layout -> A-layout transpose of P through per-wave LDS
    unsigned short* pw = &Plds[wave][0];
#pragma unroll
    for (int kt = 0; kt < 4; ++kt) {
      const int key = kt * 16 + lrow;
#pragma unroll
      for (int r = 0; r < 8; ++r)
        pw[(r + hi * 8) * KPAD + key] = f2bf(s[kt][r]);
    }
    // per-wave buffer: LDS ops from one wave are in order -> wave-local wait
    asm volatile("s_wait_dscnt 0" ::: "memory");

    // O += P @ V : 2 key-chunks x 4 d tiles
    const unsigned short* pr = &Plds[wave][0];
#pragma unroll
    for (int c = 0; c < 2; ++c) {
      Frag pf;
      const int pkb = c * 32 + hi * 8;
      const U4* pp = (const U4*)&pr[lrow * KPAD + pkb];
      pf.u[0] = pp[0];
      pf.u[1] = pp[2];
#pragma unroll
      for (int dt = 0; dt < 4; ++dt) {
        Frag vf;
        const U4* vp = (const U4*)&Vlds[cur][(dt * 16 + lrow) * KPAD + c * 32 + hi * 8];
        vf.u[0] = vp[0]; vf.u[1] = vp[2];
        oacc[dt] = wmma_bf16(pf, vf, oacc[dt]);
      }
    }

    if (has_next) {
      store_kv(1 - cur);
      __syncthreads();
    }
  }

  // epilogue: normalize and store y as bf16 [B,T,C]
  const int b = bh >> 4;
  const int h = bh & 15;
#pragma unroll
  for (int r = 0; r < 8; ++r) {
    const float inv = 1.0f / lstate[r];
    const int row = qbase + wave * 16 + r + hi * 8;
#pragma unroll
    for (int dt = 0; dt < 4; ++dt) {
      const int cidx = h * 64 + dt * 16 + lrow;
      Yb[((size_t)b * 2048 + row) * 1024 + cidx] = f2bf(oacc[dt][r] * inv);
    }
  }
}

// ---------------------------------------------------------------------------
extern "C" void kernel_launch(void* const* d_in, const int* in_sizes, int n_in,
                              void* d_out, int out_size, void* d_ws, size_t ws_size,
                              hipStream_t stream) {
  const float* x      = (const float*)d_in[0];
  const float* W_attn = (const float*)d_in[1];
  const float* b_attn = (const float*)d_in[2];
  const float* W_proj = (const float*)d_in[3];
  const float* b_proj = (const float*)d_in[4];
  float* out = (float*)d_out;

  const int B = 4, T = 2048, C = 1024, H = 16, D = 64;
  const int M  = B * T;    // 8192
  const int N1 = 3 * C;    // 3072

  unsigned char* ws = (unsigned char*)d_ws;
  size_t off = 0;
  auto take = [&](size_t bytes) -> unsigned char* {
    unsigned char* p = ws + off;
    off += (bytes + 255) & ~(size_t)255;
    return p;
  };

  unsigned short* xb  = (unsigned short*)take((size_t)M * C * 2);    // x bf16
  unsigned short* wab = (unsigned short*)take((size_t)C * N1 * 2);   // W_attn bf16
  unsigned short* wpb = (unsigned short*)take((size_t)C * C * 2);    // W_proj bf16
  unsigned short* qb  = (unsigned short*)take((size_t)B * H * T * D * 2);
  unsigned short* kbf = (unsigned short*)take((size_t)B * H * T * D * 2);
  unsigned short* vtb = (unsigned short*)take((size_t)B * H * T * D * 2);
  unsigned short* yb  = xb;   // x is dead after QKV GEMM; reuse for attention out

  cvt_kernel<<<dim3((M * C / 4 + 255) / 256), 256, 0, stream>>>(x, xb, M * C / 4);
  cvt_kernel<<<dim3((C * N1 / 4 + 255) / 256), 256, 0, stream>>>(W_attn, wab, C * N1 / 4);
  cvt_kernel<<<dim3((C * C / 4 + 255) / 256), 256, 0, stream>>>(W_proj, wpb, C * C / 4);

  // QKV projection -> Q(scaled)/K [B,H,T,D], V^T [B,H,D,T]
  gemm_kernel<1><<<dim3(N1 / 128, M / 128), 256, 0, stream>>>(
      xb, wab, b_attn, M, N1, C, qb, kbf, vtb, nullptr);

  // causal flash attention -> y bf16 [B,T,C]
  fa_kernel<<<dim3(T / 64, B * H), 128, 0, stream>>>(qb, kbf, vtb, yb);

  // output projection -> fp32 out
  gemm_kernel<0><<<dim3(C / 128, M / 128), 256, 0, stream>>>(
      yb, wpb, b_proj, M, C, C, nullptr, nullptr, nullptr, out);

  (void)in_sizes; (void)n_in; (void)out_size; (void)ws_size;
}
